// TransformerEncoderLayer_28338194219555
// MI455X (gfx1250) — matrix operand order repaired and checked by reading, not compiled, not run on hardware
//
#include <hip/hip_runtime.h>
#include <hip/hip_bf16.h>
#include <stdint.h>

#define D_MODELc 1024
#define NHEADc   16
#define D_Kc     64
#define D_FFc    4096
#define BATCHc   4
#define SEQc     2048
#define NROWSc   (BATCHc * SEQc)

typedef __attribute__((ext_vector_type(16))) __bf16 v16bf;
typedef __attribute__((ext_vector_type(8)))  float  v8f;

union FragBF { v16bf v; unsigned short u[16]; uint4 q[2]; };
union FragF  { v8f   v; float e[8]; };

__device__ __forceinline__ unsigned short f2bf(float f) {
    unsigned int u = __float_as_uint(f);
    return (unsigned short)((u + 0x7FFFu + ((u >> 16) & 1u)) >> 16);
}

// ---------------------------------------------------------------------------
// fp32 -> bf16, 4 elements/thread, vectorized
// ---------------------------------------------------------------------------
__global__ __launch_bounds__(256)
void cvt_f32_bf16(const float* __restrict__ in, unsigned short* __restrict__ out, int n4) {
    int i = blockIdx.x * 256 + threadIdx.x;
    if (i < n4) {
        float4 v = ((const float4*)in)[i];
        uint2 p;
        p.x = (unsigned)f2bf(v.x) | ((unsigned)f2bf(v.y) << 16);
        p.y = (unsigned)f2bf(v.z) | ((unsigned)f2bf(v.w) << 16);
        ((uint2*)out)[i] = p;
    }
}

// ---------------------------------------------------------------------------
// Tiled bf16 WMMA GEMM: C[M,N] = A[M,K]*B[K,N] + bias, optional ReLU.
// 128x128x32 block, 256 threads = 8 waves, 8 wmma tiles (32x64) per wave.
// LDS: A as [m][k] pitch 40, B transposed [n][k] pitch 40 -> all fragments
// are contiguous 16B runs -> ds_load_b128. Software-pipelined global loads.
// ---------------------------------------------------------------------------
#define BM 128
#define BN 128
#define BKK 32
#define APITCH 40   // elems; 80B rows, 16B aligned, conflict-free-ish
#define BPITCH 40

__global__ __launch_bounds__(256)
void gemm_bf16_wmma(const unsigned short* __restrict__ A,   // [M,K]
                    const unsigned short* __restrict__ B,   // [K,N]
                    const float* __restrict__ bias,
                    float* __restrict__ outF,
                    unsigned short* __restrict__ outH,
                    int M, int N, int K, int relu)
{
    __shared__ unsigned short As[BM * APITCH];   // [m][k]
    __shared__ unsigned short Bst[BN * BPITCH];  // [n][k] (transposed)

    const int tid  = threadIdx.x;
    const int wave = tid >> 5;
    const int lane = tid & 31;
    const int l    = lane & 15;
    const int hi   = lane >> 4;

    const int rowBase = blockIdx.x * BM;
    const int colBase = blockIdx.y * BN;

    const int wm = wave & 3;   // 32-row sub-block
    const int wn = wave >> 2;  // 64-col sub-block

    FragF acc[2][4];
#pragma unroll
    for (int mi = 0; mi < 2; ++mi)
#pragma unroll
        for (int n = 0; n < 4; ++n)
#pragma unroll
            for (int i = 0; i < 8; ++i) acc[mi][n].e[i] = 0.f;

    // staging registers for the pipelined tile
    uint4 ra0, ra1, rb0, rb1;
    // A fill indices: 512 uint4 (128 rows x 4), 2 per thread
    const int ar0 = tid >> 2,          ac0 = (tid & 3) * 8;
    const int ar1 = (tid + 256) >> 2,  ac1 = ((tid + 256) & 3) * 8;
    // B fill indices: 512 uint4 (32 k-rows x 16), 2 per thread
    const int bk0 = tid >> 4,          bn0 = (tid & 15) * 8;
    const int bk1 = (tid + 256) >> 4,  bn1 = ((tid + 256) & 15) * 8;

    auto loadTile = [&](int k0) {
        ra0 = *(const uint4*)(A + (size_t)(rowBase + ar0) * K + k0 + ac0);
        ra1 = *(const uint4*)(A + (size_t)(rowBase + ar1) * K + k0 + ac1);
        rb0 = *(const uint4*)(B + (size_t)(k0 + bk0) * N + colBase + bn0);
        rb1 = *(const uint4*)(B + (size_t)(k0 + bk1) * N + colBase + bn1);
    };
    auto storeTile = [&]() {
        ((uint4*)(As + ar0 * APITCH))[ac0 >> 3] = ra0;
        ((uint4*)(As + ar1 * APITCH))[ac1 >> 3] = ra1;
        const unsigned short* u0 = (const unsigned short*)&rb0;
        const unsigned short* u1 = (const unsigned short*)&rb1;
#pragma unroll
        for (int j = 0; j < 8; ++j) {
            Bst[(bn0 + j) * BPITCH + bk0] = u0[j];
            Bst[(bn1 + j) * BPITCH + bk1] = u1[j];
        }
    };

    loadTile(0);
    for (int k0 = 0; k0 < K; k0 += BKK) {
        __syncthreads();           // previous tile's LDS reads done
        storeTile();
        __syncthreads();
        if (k0 + BKK < K) loadTile(k0 + BKK);   // overlap with compute

        FragBF af[2];
#pragma unroll
        for (int mi = 0; mi < 2; ++mi) {
            const uint4* Arow = (const uint4*)(As + ((wm * 2 + mi) * 16 + l) * APITCH);
            af[mi].q[0] = Arow[hi];       // k = 8*hi + 0..7
            af[mi].q[1] = Arow[2 + hi];   // k = 16 + 8*hi + 0..7
        }
#pragma unroll
        for (int n = 0; n < 4; ++n) {
            FragBF bf;
            const uint4* Brow = (const uint4*)(Bst + ((wn * 4 + n) * 16 + l) * BPITCH);
            bf.q[0] = Brow[hi];           // k = 8*hi + 0..7
            bf.q[1] = Brow[2 + hi];       // k = 16 + 8*hi + 0..7
            acc[0][n].v = __builtin_amdgcn_wmma_f32_16x16x32_bf16(
                              false, af[0].v, false, bf.v, (short)0, acc[0][n].v, false, false);
            acc[1][n].v = __builtin_amdgcn_wmma_f32_16x16x32_bf16(
                              false, af[1].v, false, bf.v, (short)0, acc[1][n].v, false, false);
        }
    }

#pragma unroll
    for (int n = 0; n < 4; ++n) {
        const int gn = colBase + (wn * 4 + n) * 16 + l;
        const float bv = bias ? bias[gn] : 0.f;
#pragma unroll
        for (int mi = 0; mi < 2; ++mi) {
            const int gmBase = rowBase + (wm * 2 + mi) * 16 + hi * 8;
#pragma unroll
            for (int r = 0; r < 8; ++r) {
                float v = acc[mi][n].e[r] + bv;
                if (relu) v = fmaxf(v, 0.f);
                const size_t idx = (size_t)(gmBase + r) * N + gn;
                if (outF) outF[idx] = v;
                if (outH) outH[idx] = f2bf(v);
            }
        }
    }
}

// ---------------------------------------------------------------------------
// Flash attention (bf16 Q,K,V packed [B,S,H,64] == [N, D_MODEL]).
// 128 threads = 4 waves; wave owns a 16-row query tile. 64-key K/V tiles in
// LDS (K as [key][dk], V transposed [dk][key], pitch 72). Online softmax.
// ---------------------------------------------------------------------------
#define KPITCH 72

__global__ __launch_bounds__(128)
void attn_flash(const unsigned short* __restrict__ Qb,
                const unsigned short* __restrict__ Kb,
                const unsigned short* __restrict__ Vb,
                unsigned short* __restrict__ Ob)
{
    __shared__ unsigned short Ks[64 * KPITCH];      // [key][dk]
    __shared__ unsigned short Vt[64 * KPITCH];      // [dk][key]
    __shared__ unsigned short Ps[4 * 16 * KPITCH];  // per-wave [qrow][key]

    const int tid  = threadIdx.x;
    const int wave = tid >> 5;
    const int lane = tid & 31;
    const int l    = lane & 15;
    const int hi   = lane >> 4;

    const int bh = blockIdx.y;
    const int b  = bh / NHEADc;
    const int h  = bh % NHEADc;
    const size_t baseBH = ((size_t)b * SEQc) * D_MODELc + (size_t)h * D_Kc;
    const int q0 = blockIdx.x * 64 + wave * 16;

    // Q fragments (A-matrix), contiguous 16B runs straight from global
    FragBF qf[2];
    {
        const unsigned short* qrow = Qb + baseBH + (size_t)(q0 + l) * D_MODELc;
#pragma unroll
        for (int s = 0; s < 2; ++s) {
            qf[s].q[0] = *(const uint4*)(qrow + s * 32 + 8 * hi);
            qf[s].q[1] = *(const uint4*)(qrow + s * 32 + 16 + 8 * hi);
        }
    }

    FragF O[4];
#pragma unroll
    for (int t = 0; t < 4; ++t)
#pragma unroll
        for (int i = 0; i < 8; ++i) O[t].e[i] = 0.f;

    float mrun[8], lrun[8];
#pragma unroll
    for (int r = 0; r < 8; ++r) { mrun[r] = -1e30f; lrun[r] = 0.f; }

    const float scale = 0.125f;

    // pipelined K/V tile staging: 512 uint4 per tensor, 4 per thread
    uint4 rk[4], rv[4];
    auto loadKV = [&](int kt) {
#pragma unroll
        for (int it = 0; it < 4; ++it) {
            int i = tid + it * 128;
            int key = i >> 3, dc = (i & 7) * 8;
            rk[it] = *(const uint4*)(Kb + baseBH + (size_t)(kt + key) * D_MODELc + dc);
            rv[it] = *(const uint4*)(Vb + baseBH + (size_t)(kt + key) * D_MODELc + dc);
        }
    };
    auto storeKV = [&]() {
#pragma unroll
        for (int it = 0; it < 4; ++it) {
            int i = tid + it * 128;
            int key = i >> 3, dc = (i & 7) * 8;
            ((uint4*)(Ks + key * KPITCH))[dc >> 3] = rk[it];
            const unsigned short* vu = (const unsigned short*)&rv[it];
#pragma unroll
            for (int j = 0; j < 8; ++j)
                Vt[(dc + j) * KPITCH + key] = vu[j];
        }
    };

    loadKV(0);
    for (int kt = 0; kt < SEQc; kt += 64) {
        __syncthreads();
        storeKV();
        __syncthreads();
        if (kt + 64 < SEQc) loadKV(kt + 64);

        // S = Q*K^T : Ks[key][dk] is exactly the B-layout [n][k]
        FragF S[4];
#pragma unroll
        for (int t = 0; t < 4; ++t) {
#pragma unroll
            for (int i = 0; i < 8; ++i) S[t].e[i] = 0.f;
            const uint4* Krow = (const uint4*)(Ks + (t * 16 + l) * KPITCH);
#pragma unroll
            for (int s = 0; s < 2; ++s) {
                FragBF bf;
                bf.q[0] = Krow[4 * s + hi];
                bf.q[1] = Krow[4 * s + 2 + hi];
                S[t].v = __builtin_amdgcn_wmma_f32_16x16x32_bf16(
                             false, qf[s].v, false, bf.v, (short)0, S[t].v, false, false);
            }
        }

        // online softmax per accumulator row (row m = r + 8*hi, 16 lanes/row)
#pragma unroll
        for (int r = 0; r < 8; ++r) {
            float mx = fmaxf(fmaxf(S[0].e[r], S[1].e[r]),
                             fmaxf(S[2].e[r], S[3].e[r])) * scale;
            mx = fmaxf(mx, __shfl_xor(mx, 1));
            mx = fmaxf(mx, __shfl_xor(mx, 2));
            mx = fmaxf(mx, __shfl_xor(mx, 4));
            mx = fmaxf(mx, __shfl_xor(mx, 8));
            float mnew = fmaxf(mrun[r], mx);
            float corr = __expf(mrun[r] - mnew);
            mrun[r] = mnew;
            float pv[4], psum = 0.f;
#pragma unroll
            for (int t = 0; t < 4; ++t) {
                float p = __expf(S[t].e[r] * scale - mnew);
                pv[t] = p; psum += p;
            }
            psum += __shfl_xor(psum, 1);
            psum += __shfl_xor(psum, 2);
            psum += __shfl_xor(psum, 4);
            psum += __shfl_xor(psum, 8);
            lrun[r] = lrun[r] * corr + psum;
#pragma unroll
            for (int t = 0; t < 4; ++t) {
                O[t].e[r] *= corr;
                Ps[wave * 16 * KPITCH + (r + hi * 8) * KPITCH + t * 16 + l] = f2bf(pv[t]);
            }
        }
        asm volatile("s_wait_dscnt 0x0" ::: "memory");  // wave-local LDS RAW

        // P fragments (A-matrix) from this wave's Ps
        FragBF pf[2];
        {
            const uint4* Prow = (const uint4*)(Ps + wave * 16 * KPITCH + l * KPITCH);
#pragma unroll
            for (int s = 0; s < 2; ++s) {
                pf[s].q[0] = Prow[4 * s + hi];
                pf[s].q[1] = Prow[4 * s + 2 + hi];
            }
        }

        // O += P*V : Vt[dk][key] is the B-layout [n][k]
#pragma unroll
        for (int t = 0; t < 4; ++t) {
            const uint4* Vrow = (const uint4*)(Vt + (t * 16 + l) * KPITCH);
#pragma unroll
            for (int s = 0; s < 2; ++s) {
                FragBF bf;
                bf.q[0] = Vrow[4 * s + hi];
                bf.q[1] = Vrow[4 * s + 2 + hi];
                O[t].v = __builtin_amdgcn_wmma_f32_16x16x32_bf16(
                             false, pf[s].v, false, bf.v, (short)0, O[t].v, false, false);
            }
        }
    }

#pragma unroll
    for (int r = 0; r < 8; ++r) {
        float inv = 1.0f / lrun[r];
        int gm = q0 + r + hi * 8;
#pragma unroll
        for (int t = 0; t < 4; ++t)
            Ob[baseBH + (size_t)gm * D_MODELc + t * 16 + l] = f2bf(O[t].e[r] * inv);
    }
}

// ---------------------------------------------------------------------------
// LayerNorm(a + b), row = 1024, one 256-thread block per row, float4 I/O.
// ---------------------------------------------------------------------------
__global__ __launch_bounds__(256)
void ln_residual(const float* __restrict__ a, const float* __restrict__ b,
                 const float* __restrict__ gamma, const float* __restrict__ beta,
                 float* __restrict__ outF, unsigned short* __restrict__ outH)
{
    __shared__ float red0[8];
    __shared__ float red1[8];

    const int row  = blockIdx.x;
    const int tid  = threadIdx.x;
    const int wave = tid >> 5;
    const int lane = tid & 31;

    float4 va = ((const float4*)(a + (size_t)row * D_MODELc))[tid];
    float4 vb = ((const float4*)(b + (size_t)row * D_MODELc))[tid];
    float x0 = va.x + vb.x, x1 = va.y + vb.y, x2 = va.z + vb.z, x3 = va.w + vb.w;
    float s  = x0 + x1 + x2 + x3;
    float s2 = x0 * x0 + x1 * x1 + x2 * x2 + x3 * x3;
#pragma unroll
    for (int off = 1; off < 32; off <<= 1) {
        s  += __shfl_xor(s, off);
        s2 += __shfl_xor(s2, off);
    }
    if (lane == 0) { red0[wave] = s; red1[wave] = s2; }
    __syncthreads();
    if (wave == 0) {
        float t1 = (lane < 8) ? red0[lane] : 0.f;
        float t2 = (lane < 8) ? red1[lane] : 0.f;
#pragma unroll
        for (int off = 1; off < 8; off <<= 1) {
            t1 += __shfl_xor(t1, off);
            t2 += __shfl_xor(t2, off);
        }
        if (lane == 0) { red0[0] = t1; red1[0] = t2; }
    }
    __syncthreads();
    const float mean = red0[0] * (1.0f / D_MODELc);
    const float var  = red1[0] * (1.0f / D_MODELc) - mean * mean;
    const float rstd = rsqrtf(var + 1e-5f);

    float4 g = ((const float4*)gamma)[tid];
    float4 be = ((const float4*)beta)[tid];
    float y0 = (x0 - mean) * rstd * g.x + be.x;
    float y1 = (x1 - mean) * rstd * g.y + be.y;
    float y2 = (x2 - mean) * rstd * g.z + be.z;
    float y3 = (x3 - mean) * rstd * g.w + be.w;
    if (outF) {
        float4 o; o.x = y0; o.y = y1; o.z = y2; o.w = y3;
        ((float4*)(outF + (size_t)row * D_MODELc))[tid] = o;
    }
    if (outH) {
        uint2 p;
        p.x = (unsigned)f2bf(y0) | ((unsigned)f2bf(y1) << 16);
        p.y = (unsigned)f2bf(y2) | ((unsigned)f2bf(y3) << 16);
        ((uint2*)(outH + (size_t)row * D_MODELc))[tid] = p;
    }
}

// ---------------------------------------------------------------------------
// host-side orchestration
// ---------------------------------------------------------------------------
extern "C" void kernel_launch(void* const* d_in, const int* in_sizes, int n_in,
                              void* d_out, int out_size, void* d_ws, size_t ws_size,
                              hipStream_t stream)
{
    (void)in_sizes; (void)n_in; (void)out_size; (void)ws_size;

    const float* src = (const float*)d_in[0];
    const float* Wq = (const float*)d_in[2];  const float* bq = (const float*)d_in[3];
    const float* Wk = (const float*)d_in[4];  const float* bk = (const float*)d_in[5];
    const float* Wv = (const float*)d_in[6];  const float* bv = (const float*)d_in[7];
    const float* Wo = (const float*)d_in[8];  const float* bo = (const float*)d_in[9];
    const float* W1 = (const float*)d_in[10]; const float* b1 = (const float*)d_in[11];
    const float* W2 = (const float*)d_in[12]; const float* b2 = (const float*)d_in[13];
    const float* g1 = (const float*)d_in[14]; const float* be1 = (const float*)d_in[15];
    const float* g2 = (const float*)d_in[16]; const float* be2 = (const float*)d_in[17];
    float* out = (float*)d_out;

    char* ws = (char*)d_ws;
    size_t off = 0;
    auto alloc = [&](size_t bytes) -> void* {
        void* p = ws + off;
        off += (bytes + 255) & ~(size_t)255;
        return p;
    };

    const size_t ND = (size_t)NROWSc * D_MODELc;
    const size_t NF = (size_t)NROWSc * D_FFc;
    const size_t DD = (size_t)D_MODELc * D_MODELc;
    const size_t DF = (size_t)D_MODELc * D_FFc;

    unsigned short* srcb = (unsigned short*)alloc(ND * 2);
    unsigned short* Wqb  = (unsigned short*)alloc(DD * 2);
    unsigned short* Wkb  = (unsigned short*)alloc(DD * 2);
    unsigned short* Wvb  = (unsigned short*)alloc(DD * 2);
    unsigned short* Wob  = (unsigned short*)alloc(DD * 2);
    unsigned short* W1b  = (unsigned short*)alloc(DF * 2);
    unsigned short* W2b  = (unsigned short*)alloc(DF * 2);
    unsigned short* Qb   = (unsigned short*)alloc(ND * 2);
    unsigned short* Kb   = (unsigned short*)alloc(ND * 2);
    unsigned short* Vb   = (unsigned short*)alloc(ND * 2);
    unsigned short* Ctxb = (unsigned short*)alloc(ND * 2);
    float*          attnF= (float*)alloc(ND * 4);
    float*          x1F  = (float*)alloc(ND * 4);
    unsigned short* x1b  = (unsigned short*)alloc(ND * 2);
    unsigned short* h1b  = (unsigned short*)alloc(NF * 2);
    float*          ff2F = (float*)alloc(ND * 4);

    auto cvt = [&](const float* in, unsigned short* o, size_t n) {
        size_t n4 = n / 4;
        cvt_f32_bf16<<<(unsigned)((n4 + 255) / 256), 256, 0, stream>>>(in, o, (int)n4);
    };
    cvt(src, srcb, ND);
    cvt(Wq, Wqb, DD); cvt(Wk, Wkb, DD); cvt(Wv, Wvb, DD); cvt(Wo, Wob, DD);
    cvt(W1, W1b, DF); cvt(W2, W2b, DF);

    dim3 gD(NROWSc / BM, D_MODELc / BN);     // 64 x 8
    gemm_bf16_wmma<<<gD, 256, 0, stream>>>(srcb, Wqb, bq, nullptr, Qb,
                                           NROWSc, D_MODELc, D_MODELc, 0);
    gemm_bf16_wmma<<<gD, 256, 0, stream>>>(srcb, Wkb, bk, nullptr, Kb,
                                           NROWSc, D_MODELc, D_MODELc, 0);
    gemm_bf16_wmma<<<gD, 256, 0, stream>>>(srcb, Wvb, bv, nullptr, Vb,
                                           NROWSc, D_MODELc, D_MODELc, 0);

    dim3 gA(SEQc / 64, BATCHc * NHEADc);     // 32 x 64
    attn_flash<<<gA, 128, 0, stream>>>(Qb, Kb, Vb, Ctxb);

    gemm_bf16_wmma<<<gD, 256, 0, stream>>>(Ctxb, Wob, bo, attnF, nullptr,
                                           NROWSc, D_MODELc, D_MODELc, 0);

    ln_residual<<<NROWSc, 256, 0, stream>>>(src, attnF, g1, be1, x1F, x1b);

    dim3 gF1(NROWSc / BM, D_FFc / BN);       // 64 x 32
    gemm_bf16_wmma<<<gF1, 256, 0, stream>>>(x1b, W1b, b1, nullptr, h1b,
                                            NROWSc, D_FFc, D_MODELc, 1);
    gemm_bf16_wmma<<<gD, 256, 0, stream>>>(h1b, W2b, b2, ff2F, nullptr,
                                           NROWSc, D_MODELc, D_FFc, 0);

    ln_residual<<<NROWSc, 256, 0, stream>>>(x1F, ff2F, g2, be2, out, nullptr);
}
